// SincNetECAMambaEncoder_74998718922821
// MI455X (gfx1250) — matrix operand order repaired and checked by reading, not compiled, not run on hardware
//
#include <hip/hip_runtime.h>
#include <hip/hip_bf16.h>
#include <math.h>

typedef __attribute__((ext_vector_type(16))) _Float16 v16h;
typedef __attribute__((ext_vector_type(8)))  float    v8f;

#define PI_F 3.14159265358979323846f

__device__ __forceinline__ float siluf(float v)  { return v / (1.f + __expf(-v)); }
__device__ __forceinline__ float sigmf(float v)  { return 1.f / (1.f + __expf(-v)); }
__device__ __forceinline__ float sincf_(float v) { return (v == 0.f) ? 1.f : (__sinf(PI_F * v) / (PI_F * v)); }

#define GSL(i, N) for (long i = blockIdx.x * (long)blockDim.x + threadIdx.x; i < (N); \
                       i += (long)gridDim.x * blockDim.x)

// ---------------------------------------------------------------------------
// WMMA fragment maps (v_wmma_f32_16x16x32_f16, wave32):
// A (16x32, ISA 7.12.2): lane l -> row l&15, half i -> k = i + 8*(l>>4) + (i>=8?8:0)
// B (32x16):             lane l -> col l&15, half i -> k = i + 8*(l>>4) + (i>=8?8:0)
// ---------------------------------------------------------------------------
__device__ __forceinline__ v16h loadA_fast(const float* __restrict__ Arow,
                                           int k0, int hi, int sAk)
{
    v16h f;
    if (sAk == 1) {                       // contiguous: 4x global_load_b128
        const float4 p0 = *reinterpret_cast<const float4*>(Arow + k0 + 8 * hi);
        const float4 p1 = *reinterpret_cast<const float4*>(Arow + k0 + 8 * hi + 4);
        const float4 p2 = *reinterpret_cast<const float4*>(Arow + k0 + 16 + 8 * hi);
        const float4 p3 = *reinterpret_cast<const float4*>(Arow + k0 + 16 + 8 * hi + 4);
        f[0]=(_Float16)p0.x; f[1]=(_Float16)p0.y; f[2]=(_Float16)p0.z; f[3]=(_Float16)p0.w;
        f[4]=(_Float16)p1.x; f[5]=(_Float16)p1.y; f[6]=(_Float16)p1.z; f[7]=(_Float16)p1.w;
        f[8]=(_Float16)p2.x; f[9]=(_Float16)p2.y; f[10]=(_Float16)p2.z; f[11]=(_Float16)p2.w;
        f[12]=(_Float16)p3.x; f[13]=(_Float16)p3.y; f[14]=(_Float16)p3.z; f[15]=(_Float16)p3.w;
    } else {
#pragma unroll
        for (int i = 0; i < 16; ++i) {
            const int ka = k0 + i + 8 * hi + ((i >= 8) ? 8 : 0);
            f[i] = (_Float16)Arow[ka * sAk];
        }
    }
    return f;
}

// ---------------------------------------------------------------------------
// Pack B (any strides) into f16 WMMA-fragment order, zero-padded in K and N.
// Layout: P[(((bz*ntiles + ntile)*nK + kblk)*32 + lane)*16 + i]
// One wave per (kblk, ntile, batch); lane-adjacent stores are n-adjacent loads.
// ---------------------------------------------------------------------------
__global__ void k_packB(const float* __restrict__ Bw, int sBk, int sBn, long bsB,
                        _Float16* __restrict__ P, int N, int K, int nK, int ntiles)
{
    const int lane = threadIdx.x & 31;
    const int hi   = lane >> 4;
    const int n    = blockIdx.y * 16 + (lane & 15);
    const int k0   = blockIdx.x * 32;
    const float* Bb = Bw + (long)blockIdx.z * bsB;
    const int nc = (n < N) ? n : (N - 1);
    const float mask = (n < N) ? 1.f : 0.f;
    const float* col = Bb + nc * sBn;
    v16h f;
#pragma unroll
    for (int i = 0; i < 16; ++i) {
        const int k  = k0 + i + 8 * hi + ((i >= 8) ? 8 : 0);
        const int kc = (k < K) ? k : (K - 1);
        f[i] = (_Float16)(col[kc * sBk] * ((k < K) ? mask : 0.f));
    }
    _Float16* dst = P + ((((long)blockIdx.z * ntiles + blockIdx.y) * nK + blockIdx.x) * 32
                         + lane) * 16;
    *reinterpret_cast<v16h*>(dst) = f;
}

// ---------------------------------------------------------------------------
// GEMM: C[m,n] = sum_k A[m,k] * B[k,n]  with B pre-packed (k_packB).
// One wave per 16x16 tile; B fragment is a single 32B v16h load per K-step.
// act: 0=none 1=silu 2=tanh 3=softplus ; biasMode: 0 none, 1 per-n, 2 per-m
// ---------------------------------------------------------------------------
__global__ void k_gemm_wmma(
    const float* __restrict__ A, int sAm, int sAk, long bsA,
    const _Float16* __restrict__ P, int nK, int ntiles,
    float*       __restrict__ C, int sCm, int sCn, long bsC,
    int M, int N, int K,
    const float* __restrict__ bias, int biasMode,
    const float* __restrict__ Rsd, int sRm, int sRn, long bsR,
    int act)
{
    const int lane = threadIdx.x & 31;
    const int lrow = lane & 15;
    const int hi   = lane >> 4;
    const int m0   = blockIdx.x * 16;
    const int n0   = blockIdx.y * 16;
    const long bz  = blockIdx.z;

    const float* Ab = A + bz * bsA;
    const _Float16* Pb = P + (((long)bz * ntiles + blockIdx.y) * nK) * 512 + lane * 16;

    const int am = m0 + lrow;
    const bool aligned = (m0 + 16 <= M) && ((K & 31) == 0);
    const int amc = (am < M) ? am : (M - 1);
    const float aMask = (am < M) ? 1.f : 0.f;
    const float* Arow = Ab + amc * sAm;

    v8f acc = {0.f, 0.f, 0.f, 0.f, 0.f, 0.f, 0.f, 0.f};

    if (aligned) {
#pragma unroll 2
        for (int kb = 0; kb < nK; ++kb) {
            const int k0 = kb * 32;
            if (kb + 1 < nK)   // gfx1250 global_prefetch of next A K-slab
                __builtin_prefetch(Arow + (k0 + 32) * sAk, 0, 3);
            const v16h af = loadA_fast(Arow, k0, hi, sAk);
            const v16h bf = *reinterpret_cast<const v16h*>(Pb + (long)kb * 512);
            acc = __builtin_amdgcn_wmma_f32_16x16x32_f16(
                      false, af, false, bf, (short)0, acc, false, false);
        }
    } else {  // clamped branch-free A (B already zero-padded by packB)
        for (int kb = 0; kb < nK; ++kb) {
            const int k0 = kb * 32;
            v16h af;
#pragma unroll
            for (int i = 0; i < 16; ++i) {
                const int ka  = k0 + i + 8 * hi + ((i >= 8) ? 8 : 0);
                const int kac = (ka < K) ? ka : (K - 1);
                af[i] = (_Float16)(Arow[kac * sAk] * ((ka < K) ? aMask : 0.f));
            }
            const v16h bf = *reinterpret_cast<const v16h*>(Pb + (long)kb * 512);
            acc = __builtin_amdgcn_wmma_f32_16x16x32_f16(
                      false, af, false, bf, (short)0, acc, false, false);
        }
    }

    const int col = n0 + lrow;
    if (col < N) {
#pragma unroll
        for (int v = 0; v < 8; ++v) {
            const int row = m0 + v + 8 * hi;
            if (row < M) {
                float val = acc[v];
                if (biasMode == 1)      val += bias[col];
                else if (biasMode == 2) val += bias[row];
                if (Rsd) val += (Rsd + bz * bsR)[row * sRm + col * sRn];
                if (act == 1)      val = siluf(val);
                else if (act == 2) val = tanhf(val);
                else if (act == 3) val = (val > 20.f) ? val : log1pf(__expf(val));
                C[bz * bsC + row * sCm + col * sCn] = val;
            }
        }
    }
}

// ---------------------------------------------------------------------------
// Sinc filter bank generation: filt[c, 0..250]
// ---------------------------------------------------------------------------
__global__ void k_sincfilt(const float* __restrict__ low_hz,
                           const float* __restrict__ band_hz,
                           float* __restrict__ filt)
{
    const int c = blockIdx.x;
    const float low  = 25.f + fabsf(low_hz[c]);
    float high = low + 25.f + fabsf(band_hz[c]);
    high = fminf(fmaxf(high, 25.f), 2500.f);
    const float band = high - low;
    for (int k = threadIdx.x; k < 251; k += blockDim.x) {
        const float n    = ((float)k - 125.f) / 5000.f;
        const float win  = 0.54f - 0.46f * __cosf(2.f * PI_F * (float)k / 250.f);
        const float argl = 2.f * low  * n * low;    // faithful: sinc(2*(f*n)*f)
        const float argh = 2.f * high * n * high;
        const float lp = 2.f * low  * sincf_(argl);
        const float hp = 2.f * high * sincf_(argh);
        filt[c * 251 + k] = (hp - lp) * win / (2.f * band);
    }
}

// ---------------------------------------------------------------------------
// SincNet front conv: stride 2, pad 125, K=251, |.|, LDS-tiled
// ---------------------------------------------------------------------------
#define SC_TILE 128
__global__ void k_sinc_conv(const float* __restrict__ x,
                            const float* __restrict__ filt,
                            float* __restrict__ out, int Tin, int Tout)
{
    __shared__ float sx[2 * SC_TILE + 256];
    __shared__ float sf[256];
    const int b  = blockIdx.z;
    const int c  = blockIdx.y;
    const int t0 = blockIdx.x * SC_TILE;
    const int tid = threadIdx.x;      // 128 threads

    for (int k = tid; k < 251; k += blockDim.x) sf[k] = filt[c * 251 + k];
    const int base = 2 * t0 - 125;
    const int span = 2 * (SC_TILE - 1) + 251;      // 505
    for (int i = tid; i < span; i += blockDim.x) {
        const int gi = base + i;
        sx[i] = (gi >= 0 && gi < Tin) ? x[(long)b * Tin + gi] : 0.f;
    }
    __syncthreads();
    const int t = t0 + tid;
    if (t < Tout) {
        float acc = 0.f;
        const float* xp = &sx[2 * tid];
#pragma unroll 4
        for (int k = 0; k < 251; ++k) acc += sf[k] * xp[k];
        out[((long)b * 64 + c) * Tout + t] = fabsf(acc);
    }
}

// ---------------------------------------------------------------------------
// GroupNorm statistics: one block per (b, group)
// ---------------------------------------------------------------------------
__global__ void k_gn_stats(const float* __restrict__ x, float* __restrict__ meanO,
                           float* __restrict__ rstdO, int C, int T, int G)
{
    const int bg = blockIdx.x;
    const int b = bg / G, g = bg % G;
    const int cpg = C / G;
    const long n = (long)cpg * T;
    const float* base = x + ((long)b * C + (long)g * cpg) * T;
    float s = 0.f, ss = 0.f;
    for (long i = threadIdx.x; i < n; i += blockDim.x) {
        const float v = base[i]; s += v; ss += v * v;
    }
    __shared__ float rs[256], rq[256];
    rs[threadIdx.x] = s; rq[threadIdx.x] = ss;
    __syncthreads();
    for (int o = blockDim.x / 2; o > 0; o >>= 1) {
        if (threadIdx.x < o) { rs[threadIdx.x] += rs[threadIdx.x + o];
                               rq[threadIdx.x] += rq[threadIdx.x + o]; }
        __syncthreads();
    }
    if (threadIdx.x == 0) {
        const float mu  = rs[0] / (float)n;
        const float var = rq[0] / (float)n - mu * mu;
        meanO[bg] = mu;
        rstdO[bg] = rsqrtf(var + 1e-5f);
    }
}

__global__ void k_gn_silu(const float* __restrict__ x, float* __restrict__ out,
                          const float* __restrict__ mean, const float* __restrict__ rstd,
                          const float* __restrict__ gamma, const float* __restrict__ beta,
                          int C, int T, int G, long Ntot)
{
    const int cpg = C / G;
    GSL(i, Ntot) {
        const long rem = i / T;
        const int c = (int)(rem % C);
        const int b = (int)(rem / C);
        const int bg = b * G + c / cpg;
        const float v = (x[i] - mean[bg]) * rstd[bg] * gamma[c] + beta[c];
        out[i] = siluf(v);
    }
}

__global__ void k_avgpool2(const float* __restrict__ x, float* __restrict__ out,
                           long BC, int Tout)
{
    GSL(i, BC * Tout) {
        const long row = i / Tout;
        const long t = i % Tout;
        const float* p = x + row * (long)(2 * Tout);
        out[i] = 0.5f * (p[2 * t] + p[2 * t + 1]);
    }
}

// ---------------------------------------------------------------------------
// Depthwise conv1d (stride / pad / dilation)
// ---------------------------------------------------------------------------
__global__ void k_dwconv(const float* __restrict__ x, const float* __restrict__ w,
                         float* __restrict__ out, int C, int Tin, int Tout,
                         int Kk, int stride, int pad, int dil, long Ntot)
{
    GSL(i, Ntot) {
        const int t = (int)(i % Tout);
        const long rem = i / Tout;
        const int c = (int)(rem % C);
        const long b = rem / C;
        const float* xb = x + (b * C + c) * (long)Tin;
        float acc = 0.f;
        for (int k = 0; k < Kk; ++k) {
            const int ti = t * stride - pad + k * dil;
            if (ti >= 0 && ti < Tin) acc += w[c * Kk + k] * xb[ti];
        }
        out[i] = acc;
    }
}

// ---------------------------------------------------------------------------
// ECA: channel attention
// ---------------------------------------------------------------------------
__global__ void k_mean_t(const float* __restrict__ x, float* __restrict__ out, int T)
{
    const long bc = blockIdx.x;
    const float* p = x + bc * (long)T;
    float s = 0.f;
    for (int t = threadIdx.x; t < T; t += blockDim.x) s += p[t];
    __shared__ float rs[256];
    rs[threadIdx.x] = s; __syncthreads();
    for (int o = blockDim.x / 2; o > 0; o >>= 1) {
        if (threadIdx.x < o) rs[threadIdx.x] += rs[threadIdx.x + o];
        __syncthreads();
    }
    if (threadIdx.x == 0) out[bc] = rs[0] / (float)T;
}

__global__ void k_eca(const float* __restrict__ x, const float* __restrict__ m,
                      const float* __restrict__ w, float* __restrict__ out,
                      int C, int T, long Ntot)
{
    GSL(i, Ntot) {
        const long rem = i / T;
        const int c = (int)(rem % C);
        const long b = rem / C;
        const float* mb = m + b * C;
        float y = w[1] * mb[c];
        if (c > 0)     y += w[0] * mb[c - 1];
        if (c < C - 1) y += w[2] * mb[c + 1];
        out[i] = x[i] * sigmf(y);
    }
}

// ---------------------------------------------------------------------------
// Mamba pieces
// ---------------------------------------------------------------------------
__global__ void k_rmsnorm(const float* __restrict__ x, const float* __restrict__ g,
                          float* __restrict__ out, long BT, int D)
{
    GSL(row, BT) {
        const float* xr = x + row * D;
        float ss = 0.f;
        for (int c = 0; c < 128; ++c) ss += xr[c] * xr[c];
        const float r = rsqrtf(ss / (float)D + 1e-5f);
        float* o = out + row * D;
        for (int c = 0; c < 128; ++c) o[c] = xr[c] * r * g[c];
    }
}

// causal depthwise conv (k=4, left pad 3) on xi = xz[:, 0:256], + bias, silu
__global__ void k_cconv_silu(const float* __restrict__ xz, const float* __restrict__ cw,
                             const float* __restrict__ cb, float* __restrict__ xc,
                             int T, long Ntot)
{
    GSL(i, Ntot) {
        const int d = (int)(i % 256);
        const long rem = i / 256;
        const int t = (int)(rem % T);
        const long b = rem / T;
        float acc = cb[d];
#pragma unroll
        for (int k = 0; k < 4; ++k) {
            const int tt = t - 3 + k;
            if (tt >= 0) acc += cw[d * 4 + k] * xz[(b * T + tt) * 512L + d];
        }
        xc[i] = siluf(acc);
    }
}

// selective scan: block = one batch element, thread = one channel d (256),
// 16 SSM states kept in registers, B/C vectors broadcast through LDS.
__global__ void k_scan(const float* __restrict__ xc, const float* __restrict__ delta,
                       const float* __restrict__ dbc, const float* __restrict__ Alog,
                       float* __restrict__ ys, int T)
{
    const int b = blockIdx.x;
    const int d = threadIdx.x;   // 256
    float A[16], h[16];
#pragma unroll
    for (int n = 0; n < 16; ++n) { A[n] = -__expf(Alog[d * 16 + n]); h[n] = 0.f; }
    __shared__ float sB[16], sC[16];
    for (int t = 0; t < T; ++t) {
        const long bt = (long)b * T + t;
        if (d < 16)       sB[d]      = dbc[bt * 40 + 8 + d];
        else if (d < 32)  sC[d - 16] = dbc[bt * 40 + 24 + (d - 16)];
        __syncthreads();
        const float dt = delta[bt * 256 + d];
        const float dx = dt * xc[bt * 256 + d];
        float y = 0.f;
#pragma unroll
        for (int n = 0; n < 16; ++n) {
            h[n] = __expf(dt * A[n]) * h[n] + dx * sB[n];
            y += h[n] * sC[n];
        }
        ys[bt * 256 + d] = y;
        __syncthreads();
    }
}

// y = (ys + xc*D) * silu(z), in place over ys
__global__ void k_gate(float* __restrict__ ys, const float* __restrict__ xc,
                       const float* __restrict__ xz, const float* __restrict__ Dp,
                       long Ntot)
{
    GSL(i, Ntot) {
        const int d = (int)(i % 256);
        const long row = i / 256;
        const float z = xz[row * 512 + 256 + d];
        ys[i] = (ys[i] + xc[i] * Dp[d]) * siluf(z);
    }
}

// ---------------------------------------------------------------------------
// Attentive statistics pooling + final linear
// ---------------------------------------------------------------------------
__global__ void k_asp_score(const float* __restrict__ t64, const float* __restrict__ w2,
                            const float* __restrict__ b2, float* __restrict__ a, long BT)
{
    GSL(row, BT) {
        float acc = b2[0];
        for (int j = 0; j < 64; ++j) acc += t64[row * 64 + j] * w2[j];
        a[row] = acc;
    }
}

__global__ void k_softmax_t(const float* __restrict__ a, float* __restrict__ w, int T)
{
    const int b = blockIdx.x;
    const float* ab = a + (long)b * T;
    float* wb = w + (long)b * T;
    __shared__ float red[256];
    float m = -3.4e38f;
    for (int t = threadIdx.x; t < T; t += 256) m = fmaxf(m, ab[t]);
    red[threadIdx.x] = m; __syncthreads();
    for (int o = 128; o > 0; o >>= 1) {
        if (threadIdx.x < o) red[threadIdx.x] = fmaxf(red[threadIdx.x], red[threadIdx.x + o]);
        __syncthreads();
    }
    m = red[0]; __syncthreads();
    float s = 0.f;
    for (int t = threadIdx.x; t < T; t += 256) s += __expf(ab[t] - m);
    red[threadIdx.x] = s; __syncthreads();
    for (int o = 128; o > 0; o >>= 1) {
        if (threadIdx.x < o) red[threadIdx.x] += red[threadIdx.x + o];
        __syncthreads();
    }
    const float inv = 1.f / red[0];
    for (int t = threadIdx.x; t < T; t += 256) wb[t] = __expf(ab[t] - m) * inv;
}

// weighted mean (mean_in==nullptr) or weighted var (mean_in given); block=(b,c)
__global__ void k_wstat(const float* __restrict__ seq, const float* __restrict__ w,
                        const float* __restrict__ mean_in, float* __restrict__ out,
                        int T, int D)
{
    const int b = blockIdx.x / D, c = blockIdx.x % D;
    float acc = 0.f;
    for (int t = threadIdx.x; t < T; t += blockDim.x) {
        const long bt = (long)b * T + t;
        const float s = seq[bt * D + c];
        if (mean_in) { const float dlt = s - mean_in[b * D + c]; acc += w[bt] * dlt * dlt; }
        else         { acc += w[bt] * s; }
    }
    __shared__ float rs[256];
    rs[threadIdx.x] = acc; __syncthreads();
    for (int o = blockDim.x / 2; o > 0; o >>= 1) {
        if (threadIdx.x < o) rs[threadIdx.x] += rs[threadIdx.x + o];
        __syncthreads();
    }
    if (threadIdx.x == 0) out[b * D + c] = rs[0];
}

__global__ void k_final(const float* __restrict__ mean, const float* __restrict__ var,
                        const float* __restrict__ cw, const float* __restrict__ cb,
                        float* __restrict__ out, int D, int O, long Ntot)
{
    GSL(i, Ntot) {
        const int o = (int)(i % O);
        const int b = (int)(i / O);
        float acc = cb[o];
        for (int c = 0; c < D; ++c) acc += mean[b * D + c] * cw[c * O + o];
        for (int c = 0; c < D; ++c) acc += sqrtf(var[b * D + c] + 1e-8f) * cw[(D + c) * O + o];
        out[i] = acc;
    }
}

// ===========================================================================
// Host side
// ===========================================================================
static inline int nblk(long n, int t = 256) {
    long b = (n + t - 1) / t;
    return (int)(b > 1048576 ? 1048576 : b);
}

// pack B then run the WMMA GEMM
static inline void gemm(hipStream_t s,
                        const float* A, int sAm, int sAk, long bsA,
                        const float* B, int sBk, int sBn, long bsB,
                        float* C, int sCm, int sCn, long bsC,
                        int M, int N, int K, int batch,
                        const float* bias, int biasMode,
                        const float* R, int sRm, int sRn, long bsR, int act,
                        _Float16* packBuf)
{
    const int nK = (K + 31) / 32;
    const int ntiles = (N + 15) / 16;
    dim3 gp(nK, ntiles, batch);
    k_packB<<<gp, 32, 0, s>>>(B, sBk, sBn, bsB, packBuf, N, K, nK, ntiles);
    dim3 g((M + 15) / 16, ntiles, batch);
    k_gemm_wmma<<<g, 32, 0, s>>>(A, sAm, sAk, bsA, packBuf, nK, ntiles,
                                 C, sCm, sCn, bsC, M, N, K,
                                 bias, biasMode, R, sRm, sRn, bsR, act);
}

extern "C" void kernel_launch(void* const* d_in, const int* in_sizes, int n_in,
                              void* d_out, int out_size, void* d_ws, size_t ws_size,
                              hipStream_t stream) {
    (void)in_sizes; (void)n_in; (void)out_size; (void)ws_size;
    const float* x        = (const float*)d_in[0];
    const float* low_hz   = (const float*)d_in[1];
    const float* band_hz  = (const float*)d_in[2];
    const float* sinc_g   = (const float*)d_in[3];
    const float* sinc_b   = (const float*)d_in[4];
    const float* dsw[4][6] = {
        {(const float*)d_in[5],  (const float*)d_in[6],  (const float*)d_in[7],
         (const float*)d_in[8],  (const float*)d_in[9],  (const float*)d_in[10]},
        {(const float*)d_in[11], (const float*)d_in[12], (const float*)d_in[13],
         (const float*)d_in[14], (const float*)d_in[15], (const float*)d_in[16]},
        {(const float*)d_in[17], (const float*)d_in[18], (const float*)d_in[19],
         (const float*)d_in[20], (const float*)d_in[21], (const float*)d_in[22]},
        {(const float*)d_in[23], (const float*)d_in[24], (const float*)d_in[25],
         (const float*)d_in[26], (const float*)d_in[27], (const float*)d_in[28]}};
    const float* eca_w   = (const float*)d_in[29];
    const float* tok_w   = (const float*)d_in[30];
    const float* tok_b   = (const float*)d_in[31];
    const float* m_norm  = (const float*)d_in[32];
    const float* m_in    = (const float*)d_in[33];
    const float* m_convw = (const float*)d_in[34];
    const float* m_convb = (const float*)d_in[35];
    const float* m_xproj = (const float*)d_in[36];
    const float* m_dtw   = (const float*)d_in[37];
    const float* m_dtb   = (const float*)d_in[38];
    const float* m_Alog  = (const float*)d_in[39];
    const float* m_D     = (const float*)d_in[40];
    const float* m_out   = (const float*)d_in[41];
    const float* asp_w1  = (const float*)d_in[42];
    const float* asp_b1  = (const float*)d_in[43];
    const float* asp_w2  = (const float*)d_in[44];
    const float* asp_b2  = (const float*)d_in[45];
    const float* cyc_w   = (const float*)d_in[46];
    const float* cyc_b   = (const float*)d_in[47];
    float* out = (float*)d_out;

    const int Bz = 16;
    const long BT = 16L * 6250;

    // ---- workspace arena (floats) ----
    float* w = (float*)d_ws;
    size_t off = 0;
    auto alloc = [&](size_t n) { size_t r = off; off += (n + 255) & ~(size_t)255; return w + r; };
    float* F_FILT = alloc(64 * 251);
    float* F_H    = alloc(16L * 64 * 25000);
    float* F_MEAN = alloc(4096);
    float* F_RSTD = alloc(4096);
    float* F_TA   = alloc(16L * 96 * 12500);
    float* F_TB   = alloc(16L * 96 * 12500);
    float* F_ECAM = alloc(4096);
    float* F_SEQ0 = alloc(BT * 128);
    float* F_SEQ1 = alloc(BT * 128);
    float* F_RMS  = alloc(BT * 128);
    float* F_XZ   = alloc(BT * 512);
    float* F_XC   = alloc(BT * 256);
    float* F_DBC  = alloc(BT * 40);
    float* F_DEL  = alloc(BT * 256);
    float* F_YS   = alloc(BT * 256);
    float* F_ASPT = alloc(BT * 64);
    float* F_ASPA = alloc(BT);
    float* F_ASPW = alloc(BT);
    float* F_PM   = alloc(2048);
    float* F_PV   = alloc(2048);
    _Float16* F_PACK = (_Float16*)alloc(7000000);   // 14M halves for packed B

    // ---- SincNet front end ----
    k_sincfilt<<<64, 256, 0, stream>>>(low_hz, band_hz, F_FILT);
    {
        dim3 g((25000 + SC_TILE - 1) / SC_TILE, 64, 16);
        k_sinc_conv<<<g, SC_TILE, 0, stream>>>(x, F_FILT, F_H, 50000, 25000);
    }
    k_gn_stats<<<16 * 8, 256, 0, stream>>>(F_H, F_MEAN, F_RSTD, 64, 25000, 8);
    {
        long N = 16L * 64 * 25000;
        k_gn_silu<<<nblk(N), 256, 0, stream>>>(F_H, F_H, F_MEAN, F_RSTD, sinc_g, sinc_b,
                                               64, 25000, 8, N);
    }
    k_avgpool2<<<nblk(16L * 64 * 12500), 256, 0, stream>>>(F_H, F_TA, 16L * 64, 12500);

    // ---- DSConv + ECA stack ----
    struct Blk { int Cin, Cout, Tin, Tout, K, stride, pad, dil; };
    const Blk blks[4] = {{64, 96, 12500, 12500, 7, 1, 3, 1},
                         {96, 96, 12500, 6250, 5, 2, 2, 1},
                         {96, 128, 6250, 6250, 5, 1, 2, 1},
                         {128, 128, 6250, 6250, 5, 1, 4, 2}};
    float* cur = F_TA;
    float* oth = F_TB;
    for (int i = 0; i < 4; ++i) {
        const Blk& bk = blks[i];
        long Ndw = 16L * bk.Cin * bk.Tout;
        k_dwconv<<<nblk(Ndw), 256, 0, stream>>>(cur, dsw[i][0], oth, bk.Cin, bk.Tin,
                                                bk.Tout, bk.K, bk.stride, bk.pad, bk.dil, Ndw);
        k_gn_stats<<<16 * 8, 256, 0, stream>>>(oth, F_MEAN, F_RSTD, bk.Cin, bk.Tout, 8);
        k_gn_silu<<<nblk(Ndw), 256, 0, stream>>>(oth, oth, F_MEAN, F_RSTD, dsw[i][2],
                                                 dsw[i][3], bk.Cin, bk.Tout, 8, Ndw);
        // pointwise conv:   C[b][co][t] = W[co][ci] * X[b][ci][t]   (WMMA)
        gemm(stream, dsw[i][1], bk.Cin, 1, 0,
             oth, bk.Tout, 1, (long)bk.Cin * bk.Tout,
             cur, bk.Tout, 1, (long)bk.Cout * bk.Tout,
             bk.Cout, bk.Tout, bk.Cin, 16, nullptr, 0, nullptr, 0, 0, 0, 0, F_PACK);
        long Npw = 16L * bk.Cout * bk.Tout;
        k_gn_stats<<<16 * 8, 256, 0, stream>>>(cur, F_MEAN, F_RSTD, bk.Cout, bk.Tout, 8);
        k_gn_silu<<<nblk(Npw), 256, 0, stream>>>(cur, cur, F_MEAN, F_RSTD, dsw[i][4],
                                                 dsw[i][5], bk.Cout, bk.Tout, 8, Npw);
        k_mean_t<<<16 * bk.Cout, 256, 0, stream>>>(cur, F_ECAM, bk.Tout);
        k_eca<<<nblk(Npw), 256, 0, stream>>>(cur, F_ECAM, eca_w + 3 * i, cur,
                                             bk.Cout, bk.Tout, Npw);
    }

    // ---- token projection -> seq [B, T, 128] (store transposed: row=co, col=t) ----
    gemm(stream, tok_w, 128, 1, 0,
         cur, 6250, 1, 128L * 6250,
         F_SEQ0, 1, 128, 6250L * 128,
         128, 6250, 128, 16, tok_b, 2, nullptr, 0, 0, 0, 0, F_PACK);

    // ---- Mamba layers ----
    float* seqIn = F_SEQ0;
    float* seqOut = F_SEQ1;
    for (int l = 0; l < 2; ++l) {
        const float* nw   = m_norm  + l * 128;
        const float* Win  = m_in    + l * 128 * 512;
        const float* cw   = m_convw + l * 256 * 4;
        const float* cb   = m_convb + l * 256;
        const float* Wx   = m_xproj + l * 256 * 40;
        const float* Wdt  = m_dtw   + l * 8 * 256;
        const float* bdt  = m_dtb   + l * 256;
        const float* Alog = m_Alog  + l * 256 * 16;
        const float* Dp   = m_D     + l * 256;
        const float* Wout = m_out   + l * 256 * 128;

        k_rmsnorm<<<nblk(BT), 256, 0, stream>>>(seqIn, nw, F_RMS, BT, 128);
        gemm(stream, F_RMS, 128, 1, 0, Win, 512, 1, 0, F_XZ, 512, 1, 0,
             (int)BT, 512, 128, 1, nullptr, 0, nullptr, 0, 0, 0, 0, F_PACK);
        k_cconv_silu<<<nblk(BT * 256), 256, 0, stream>>>(F_XZ, cw, cb, F_XC, 6250, BT * 256);
        gemm(stream, F_XC, 256, 1, 0, Wx, 40, 1, 0, F_DBC, 40, 1, 0,
             (int)BT, 40, 256, 1, nullptr, 0, nullptr, 0, 0, 0, 0, F_PACK);
        gemm(stream, F_DBC, 40, 1, 0, Wdt, 256, 1, 0, F_DEL, 256, 1, 0,
             (int)BT, 256, 8, 1, bdt, 1, nullptr, 0, 0, 0, 3 /*softplus*/, F_PACK);
        k_scan<<<16, 256, 0, stream>>>(F_XC, F_DEL, F_DBC, Alog, F_YS, 6250);
        k_gate<<<nblk(BT * 256), 256, 0, stream>>>(F_YS, F_XC, F_XZ, Dp, BT * 256);
        gemm(stream, F_YS, 256, 1, 0, Wout, 128, 1, 0, seqOut, 128, 1, 0,
             (int)BT, 128, 256, 1, nullptr, 0, seqIn, 128, 1, 0, 0, F_PACK);
        float* tmp = seqIn; seqIn = seqOut; seqOut = tmp;
    }
    float* seq = seqIn;   // after 2 swaps -> F_SEQ0

    // ---- attentive statistics pooling ----
    gemm(stream, seq, 128, 1, 0, asp_w1, 64, 1, 0, F_ASPT, 64, 1, 0,
         (int)BT, 64, 128, 1, asp_b1, 1, nullptr, 0, 0, 0, 2 /*tanh*/, F_PACK);
    k_asp_score<<<nblk(BT), 256, 0, stream>>>(F_ASPT, asp_w2, asp_b2, F_ASPA, BT);
    k_softmax_t<<<Bz, 256, 0, stream>>>(F_ASPA, F_ASPW, 6250);
    k_wstat<<<Bz * 128, 256, 0, stream>>>(seq, F_ASPW, nullptr, F_PM, 6250, 128);
    k_wstat<<<Bz * 128, 256, 0, stream>>>(seq, F_ASPW, F_PM, F_PV, 6250, 128);
    k_final<<<nblk(16L * 192), 256, 0, stream>>>(F_PM, F_PV, cyc_w, cyc_b, out,
                                                 128, 192, 16L * 192);
}
